// SoftMoE2_3719441678637
// MI455X (gfx1250) — matrix operand order repaired and checked by reading, not compiled, not run on hardware
//
#include <hip/hip_runtime.h>
#include <hip/hip_bf16.h>
#include <math.h>

// ---------------------------------------------------------------------------
// CDNA5 / gfx1250 SoftMoE forward.  Five large GEMMs on
// v_wmma_f32_16x16x32_bf16 (bf16 A/B, fp32 accumulate), with double-buffered
// LDS tiles filled by GLOBAL_LOAD_ASYNC_TO_LDS_B128 (ASYNCcnt path).
// ---------------------------------------------------------------------------

typedef __attribute__((ext_vector_type(4)))  unsigned int u32x4;
typedef __attribute__((ext_vector_type(8)))  float        f32x8;
typedef __attribute__((ext_vector_type(16))) __bf16       bf16x16;

union FragAB { bf16x16 v; u32x4 q[2]; };        // 32 bytes: one WMMA A or B operand
union Pack8  { u32x4 q; __bf16 e[8]; };

#define BLK_M 128
#define BLK_N 128
#define BLK_K 32

__device__ __forceinline__ float gelu_exact(float v) {
    return 0.5f * v * (1.0f + erff(v * 0.70710678118654752440f));
}

// Async DMA of 16 bytes global -> LDS (no VGPR staging, tracked by ASYNCcnt).
// LDS byte address = low 32 bits of the flat shared pointer.
__device__ __forceinline__ void async_copy_b128(void* lds, const void* gptr) {
    unsigned loff = (unsigned)(unsigned long long)lds;
    asm volatile("global_load_async_to_lds_b128 %0, %1, off"
                 :: "v"(loff), "v"(gptr) : "memory");
}
__device__ __forceinline__ void wait_async0() {
    asm volatile("s_wait_asynccnt 0x0" ::: "memory");
}

// ---------------------------------------------------------------------------
// Tiled bf16 WMMA GEMM:  C[M,N] = A[M,K] * B
//   B_IS_NK = true : B stored as [N,K] rows (multiply by B^T; async staged)
//   B_IS_NK = false: B stored as [K,N] row-major, transposed into LDS on load
//   EPI: 0 = fp32 store, 1 = bf16 store, 2 = exact GELU then bf16 store
// 256 threads = 8 waves (wave32), 2x4 wave grid, each wave owns 64x32 output.
// Double-buffered LDS: stage tile k+1 async while tile k feeds the WMMAs.
// ---------------------------------------------------------------------------
template<bool B_IS_NK, int EPI>
__global__ __launch_bounds__(256)
void gemm_bf16_wmma(const __bf16* __restrict__ A, const __bf16* __restrict__ B,
                    void* __restrict__ C,
                    int M, int N, int K, int lda, int ldb, int ldc,
                    long long sAb, long long sBb, long long sCb)
{
    __shared__ __bf16 lA[2][BLK_M][BLK_K];   // 2 x 8 KB
    __shared__ __bf16 lB[2][BLK_N][BLK_K];   // 2 x 8 KB, stored [n][k]

    A += (long long)blockIdx.z * sAb;
    B += (long long)blockIdx.z * sBb;

    const int tid  = threadIdx.x;
    const int lane = tid & 31;
    const int wave = tid >> 5;
    const int wr   = wave & 1;     // 2 wave-rows of 64 in M
    const int wc   = wave >> 1;    // 4 wave-cols of 32 in N
    const int m0   = blockIdx.y * BLK_M;
    const int n0   = blockIdx.x * BLK_N;
    const int lr   = lane & 15;
    const int hi   = lane >> 4;

    f32x8 acc[4][2];
    #pragma unroll
    for (int mt = 0; mt < 4; ++mt)
        #pragma unroll
        for (int nt = 0; nt < 2; ++nt)
            #pragma unroll
            for (int r = 0; r < 8; ++r) acc[mt][nt][r] = 0.0f;

    // ---- tile stager: A (and B when [N,K]) via async DMA; [K,N] B via LDS transpose
    auto stage = [&](int buf, int k0) {
        {
            const int m  = tid >> 2;            // 0..63
            const int kk = (tid & 3) * 8;       // 0,8,16,24
            async_copy_b128(&lA[buf][m][kk],      &A[(size_t)(m0 + m)      * lda + k0 + kk]);
            async_copy_b128(&lA[buf][m + 64][kk], &A[(size_t)(m0 + m + 64) * lda + k0 + kk]);
        }
        if (B_IS_NK) {
            const int n  = tid >> 2;
            const int kk = (tid & 3) * 8;
            async_copy_b128(&lB[buf][n][kk],      &B[(size_t)(n0 + n)      * ldb + k0 + kk]);
            async_copy_b128(&lB[buf][n + 64][kk], &B[(size_t)(n0 + n + 64) * ldb + k0 + kk]);
        } else {
            const int k = tid >> 4;             // 0..15
            const int n = (tid & 15) * 8;       // 0..120
            #pragma unroll
            for (int kk = 0; kk < BLK_K; kk += 16) {
                Pack8 r;
                r.q = *(const u32x4*)&B[(size_t)(k0 + k + kk) * ldb + n0 + n];
                #pragma unroll
                for (int i = 0; i < 8; ++i) lB[buf][n + i][k + kk] = r.e[i];
            }
        }
    };

    const int KT = K / BLK_K;
    stage(0, 0);
    wait_async0();
    __syncthreads();

    for (int kt = 0; kt < KT; ++kt) {
        const int buf = kt & 1;

        // kick off the next tile into the other buffer (overlaps the WMMAs)
        if (kt + 1 < KT) stage(buf ^ 1, (kt + 1) * BLK_K);
        // near-cache prefetch two tiles ahead on the A stream
        if (kt + 2 < KT)
            __builtin_prefetch(&A[(size_t)(m0 + (tid >> 2)) * lda
                                  + (kt + 2) * BLK_K + (tid & 3) * 8], 0, 3);

        // ---- fragments per ISA VGPR layouts ---------------------------
        // A 16x32 bf16: lane(lr,hi): e[0..7]=K hi*8+0..7, e[8..15]=K 16+hi*8+0..7
        FragAB af[4];
        #pragma unroll
        for (int mt = 0; mt < 4; ++mt) {
            const int r = wr * 64 + mt * 16 + lr;
            af[mt].q[0] = *(const u32x4*)&lA[buf][r][hi * 8];
            af[mt].q[1] = *(const u32x4*)&lA[buf][r][16 + hi * 8];
        }
        // B 32x16 bf16: lane(lr,hi): e[0..7]=K hi*8+0..7, e[8..15]=K 16+hi*8+0..7 of column lr
        FragAB bfr[2];
        #pragma unroll
        for (int nt = 0; nt < 2; ++nt) {
            const int c = wc * 32 + nt * 16 + lr;
            bfr[nt].q[0] = *(const u32x4*)&lB[buf][c][hi * 8];
            bfr[nt].q[1] = *(const u32x4*)&lB[buf][c][16 + hi * 8];
        }

        #pragma unroll
        for (int mt = 0; mt < 4; ++mt)
            #pragma unroll
            for (int nt = 0; nt < 2; ++nt)
                acc[mt][nt] = __builtin_amdgcn_wmma_f32_16x16x32_bf16(
                    false, af[mt].v, false, bfr[nt].v,
                    (short)0, acc[mt][nt], false, false);

        // publish the freshly staged buffer workgroup-wide
        wait_async0();
        __syncthreads();
    }

    // ---- epilogue: C/D layout -> lane lr = column, VGPR r = row r+8*hi ----
    if (EPI == 0) {
        float* Cp = (float*)C + (size_t)blockIdx.z * (size_t)sCb;
        #pragma unroll
        for (int mt = 0; mt < 4; ++mt)
            #pragma unroll
            for (int nt = 0; nt < 2; ++nt) {
                const int row0 = m0 + wr * 64 + mt * 16 + hi * 8;
                const int col  = n0 + wc * 32 + nt * 16 + lr;
                #pragma unroll
                for (int r = 0; r < 8; ++r)
                    Cp[(size_t)(row0 + r) * ldc + col] = acc[mt][nt][r];
            }
    } else {
        __bf16* Cp = (__bf16*)C + (size_t)blockIdx.z * (size_t)sCb;
        #pragma unroll
        for (int mt = 0; mt < 4; ++mt)
            #pragma unroll
            for (int nt = 0; nt < 2; ++nt) {
                const int row0 = m0 + wr * 64 + mt * 16 + hi * 8;
                const int col  = n0 + wc * 32 + nt * 16 + lr;
                #pragma unroll
                for (int r = 0; r < 8; ++r) {
                    float v = acc[mt][nt][r];
                    if (EPI == 2) v = gelu_exact(v);
                    Cp[(size_t)(row0 + r) * ldc + col] = (__bf16)v;
                }
            }
    }
}

// ---------------------------------------------------------------------------
// RMSNorm (l2norm * sqrt(d) * gamma), fp32 in -> bf16 out.  One block per row.
// ---------------------------------------------------------------------------
__global__ __launch_bounds__(256)
void rmsnorm_to_bf16(const float* __restrict__ x, const float* __restrict__ gamma,
                     __bf16* __restrict__ out, int d)
{
    __shared__ float red[256];
    const float* xr = x + (size_t)blockIdx.x * d;
    __bf16* orow    = out + (size_t)blockIdx.x * d;
    float ss = 0.0f;
    for (int i = threadIdx.x; i < d; i += 256) { float v = xr[i]; ss += v * v; }
    red[threadIdx.x] = ss; __syncthreads();
    for (int s = 128; s > 0; s >>= 1) {
        if (threadIdx.x < s) red[threadIdx.x] += red[threadIdx.x + s];
        __syncthreads();
    }
    const float n     = fmaxf(sqrtf(red[0]), 1e-12f);
    const float scale = sqrtf((float)d) / n;
    for (int i = threadIdx.x; i < d; i += 256)
        orow[i] = (__bf16)(xr[i] * scale * gamma[i]);
}

// fp32 -> bf16 elementwise
__global__ __launch_bounds__(256)
void f2bf(const float* __restrict__ in, __bf16* __restrict__ out, int n)
{
    int i = blockIdx.x * 256 + threadIdx.x;
    if (i < n) out[i] = (__bf16)in[i];
}

// ---------------------------------------------------------------------------
// Dispatch softmax (over s): online column max/sum.  One thread per column.
// ---------------------------------------------------------------------------
__global__ __launch_bounds__(256)
void col_softmax_stats(const float* __restrict__ logits,
                       float* __restrict__ cmax, float* __restrict__ csum,
                       int S, int N)
{
    const int n = blockIdx.x * 256 + threadIdx.x;
    const float* L = logits + (size_t)blockIdx.z * S * N;
    float m = -3.0e38f, s = 0.0f;
    for (int i = 0; i < S; ++i) {
        const float x  = L[(size_t)i * N + n];
        const float mn = fmaxf(m, x);
        s = s * __expf(m - mn) + __expf(x - mn);
        m = mn;
    }
    cmax[(size_t)blockIdx.z * N + n] = m;
    csum[(size_t)blockIdx.z * N + n] = s;
}

// Build dispatch^T bf16 [b][n][s] with a 64x64 LDS transpose (coalesced both ways).
__global__ __launch_bounds__(256)
void build_dispT(const float* __restrict__ logits,
                 const float* __restrict__ cmax, const float* __restrict__ csum,
                 __bf16* __restrict__ dispT, int S, int N)
{
    __shared__ float tile[64][65];
    __shared__ float cm[64], ci[64];
    const size_t boff = (size_t)blockIdx.z * S * N;
    const float* L = logits + boff;
    __bf16* Dp     = dispT  + boff;
    const int s0 = blockIdx.x * 64, n0 = blockIdx.y * 64;
    if (threadIdx.x < 64) {
        cm[threadIdx.x] = cmax[(size_t)blockIdx.z * N + n0 + threadIdx.x];
        ci[threadIdx.x] = 1.0f / csum[(size_t)blockIdx.z * N + n0 + threadIdx.x];
    }
    __syncthreads();
    for (int idx = threadIdx.x; idx < 64 * 64; idx += 256) {
        const int i = idx >> 6, j = idx & 63;
        const float v = L[(size_t)(s0 + i) * N + n0 + j];
        tile[i][j] = __expf(v - cm[j]) * ci[j];
    }
    __syncthreads();
    for (int idx = threadIdx.x; idx < 64 * 64; idx += 256) {
        const int j = idx >> 6, i = idx & 63;
        Dp[(size_t)(n0 + j) * S + s0 + i] = (__bf16)tile[i][j];
    }
}

// Combine softmax (over e*t): row softmax, fp32 logits -> bf16.  Block per row.
__global__ __launch_bounds__(256)
void row_softmax_bf16(const float* __restrict__ logits, __bf16* __restrict__ comb, int N)
{
    __shared__ float red[256];
    const float* L = logits + (size_t)blockIdx.x * N;
    __bf16* O      = comb   + (size_t)blockIdx.x * N;
    float m = -3.0e38f;
    for (int i = threadIdx.x; i < N; i += 256) m = fmaxf(m, L[i]);
    red[threadIdx.x] = m; __syncthreads();
    for (int s = 128; s > 0; s >>= 1) {
        if (threadIdx.x < s) red[threadIdx.x] = fmaxf(red[threadIdx.x], red[threadIdx.x + s]);
        __syncthreads();
    }
    m = red[0]; __syncthreads();
    float acc = 0.0f;
    for (int i = threadIdx.x; i < N; i += 256) acc += __expf(L[i] - m);
    red[threadIdx.x] = acc; __syncthreads();
    for (int s = 128; s > 0; s >>= 1) {
        if (threadIdx.x < s) red[threadIdx.x] += red[threadIdx.x + s];
        __syncthreads();
    }
    const float inv = 1.0f / red[0];
    for (int i = threadIdx.x; i < N; i += 256)
        O[i] = (__bf16)(__expf(L[i] - m) * inv);
}

// ---------------------------------------------------------------------------
extern "C" void kernel_launch(void* const* d_in, const int* in_sizes, int n_in,
                              void* d_out, int out_size, void* d_ws, size_t ws_size,
                              hipStream_t stream)
{
    (void)in_sizes; (void)n_in; (void)out_size; (void)ws_size;

    const int Bb = 4, S = 4096, D = 1024, E = 8, T = 512;
    const int Ns = E * T;        // 4096 slots
    const int H  = 4 * D;        // 4096 hidden

    const float* x  = (const float*)d_in[0];
    const float* gx = (const float*)d_in[1];
    const float* gs = (const float*)d_in[2];
    const float* se = (const float*)d_in[3];
    const float* w1 = (const float*)d_in[4];
    const float* w2 = (const float*)d_in[5];
    float* out = (float*)d_out;

    // ---- workspace partition (256B aligned) ----
    char* p = (char*)d_ws;
    auto alloc = [&](size_t bytes) {
        char* r = p; p += (bytes + 255) & ~(size_t)255; return (void*)r;
    };
    __bf16* xn     = (__bf16*)alloc((size_t)Bb * S * D * 2);   // rmsnorm(x)
    __bf16* seb    = (__bf16*)alloc((size_t)Ns * D * 2);       // rmsnorm(slot_embeds)
    __bf16* w1b    = (__bf16*)alloc((size_t)D * H * 2);
    __bf16* w2b    = (__bf16*)alloc((size_t)H * D * 2);
    float*  logits = (float*) alloc((size_t)Bb * S * Ns * 4);  // kept for both softmaxes
    float*  cmax   = (float*) alloc((size_t)Bb * Ns * 4);
    float*  csum   = (float*) alloc((size_t)Bb * Ns * 4);
    // 128 MB region reused serially: dispatch^T -> h -> combine
    __bf16* big    = (__bf16*)alloc((size_t)Bb * S * Ns * 2);
    __bf16* slots  = (__bf16*)alloc((size_t)Bb * Ns * D * 2);
    __bf16* proc   = (__bf16*)alloc((size_t)Bb * Ns * D * 2);
    __bf16* dispT   = big;
    __bf16* hbuf    = big;
    __bf16* combine = big;

    // ---- 1) normalizations + weight casts ----
    rmsnorm_to_bf16<<<Bb * S, 256, 0, stream>>>(x,  gx, xn,  D);
    rmsnorm_to_bf16<<<Ns,     256, 0, stream>>>(se, gs, seb, D);
    f2bf<<<(D * H + 255) / 256, 256, 0, stream>>>(w1, w1b, D * H);
    f2bf<<<(H * D + 255) / 256, 256, 0, stream>>>(w2, w2b, H * D);

    // ---- 2) logits[b] = xn[b] @ seb^T   (fp32 out) ----
    {
        dim3 g(Ns / BLK_N, S / BLK_M, Bb);
        gemm_bf16_wmma<true, 0><<<g, 256, 0, stream>>>(
            xn, seb, logits, S, Ns, D, D, D, Ns,
            (long long)S * D, 0LL, (long long)S * Ns);
    }

    // ---- 3) dispatch softmax over s, emit dispatch^T bf16 [b][n][s] ----
    col_softmax_stats<<<dim3(Ns / 256, 1, Bb), 256, 0, stream>>>(logits, cmax, csum, S, Ns);
    build_dispT<<<dim3(S / 64, Ns / 64, Bb), 256, 0, stream>>>(logits, cmax, csum, dispT, S, Ns);

    // ---- 4) slots[b] = dispatch^T[b] @ xn[b]   (bf16 out) ----
    {
        dim3 g(D / BLK_N, Ns / BLK_M, Bb);
        gemm_bf16_wmma<false, 1><<<g, 256, 0, stream>>>(
            dispT, xn, slots, Ns, D, S, S, D, D,
            (long long)Ns * S, (long long)S * D, (long long)Ns * D);
    }

    // ---- 5) h = gelu(slots @ w1)  [B*Ns, H] (bf16 out) ----
    {
        dim3 g(H / BLK_N, (Bb * Ns) / BLK_M, 1);
        gemm_bf16_wmma<false, 2><<<g, 256, 0, stream>>>(
            slots, w1b, hbuf, Bb * Ns, H, D, D, H, H, 0LL, 0LL, 0LL);
    }

    // ---- 6) processed = h @ w2  [B*Ns, D] (bf16 out) ----
    {
        dim3 g(D / BLK_N, (Bb * Ns) / BLK_M, 1);
        gemm_bf16_wmma<false, 1><<<g, 256, 0, stream>>>(
            hbuf, w2b, proc, Bb * Ns, D, H, H, D, D, 0LL, 0LL, 0LL);
    }

    // ---- 7) combine softmax over slots (row-wise), bf16 ----
    row_softmax_bf16<<<Bb * S, 256, 0, stream>>>(logits, combine, Ns);

    // ---- 8) out[b] = combine[b] @ processed[b]   (fp32 out) ----
    {
        dim3 g(D / BLK_N, S / BLK_M, Bb);
        gemm_bf16_wmma<false, 0><<<g, 256, 0, stream>>>(
            combine, proc, out, S, D, Ns, Ns, D, D,
            (long long)S * Ns, (long long)Ns * D, (long long)S * D);
    }
}
